// CausalSelfAttention_14731737825808
// MI455X (gfx1250) — matrix operand order repaired and checked by reading, not compiled, not run on hardware
//
#include <hip/hip_runtime.h>
#include <math.h>

#define BB 4
#define SS 2048
#define DD 1024
#define HH 16
#define HDIM 64
#define SCALE_Q 0.125f  // 1/sqrt(64)

typedef __bf16 bf16_t;
typedef __attribute__((ext_vector_type(16))) __bf16 v16bf;
typedef __attribute__((ext_vector_type(8)))  __bf16 v8bf;
typedef __attribute__((ext_vector_type(8)))  float  v8f;

// ---------- helpers ----------

__device__ __forceinline__ bf16_t f2bf(float f) {
  union { float f; unsigned u; } v; v.f = f;
  unsigned r = v.u + 0x7FFFu + ((v.u >> 16) & 1u);  // round-to-nearest-even
  unsigned short h = (unsigned short)(r >> 16);
  bf16_t b;
  __builtin_memcpy(&b, &h, sizeof(b));
  return b;
}

// A-operand fragment (16xK tile, row-major, ld = row stride in elements).
// ISA 7.12.2 16-bit A layout: lanes 0-15 -> row M=lane, halves 0..7 = K 0..7,
// halves 8..15 = K 16..23; lanes 16-31 -> same row, K 8..15 / 24..31.
__device__ __forceinline__ v16bf load_frag_a(const bf16_t* __restrict__ base, int ld) {
  const int lane = threadIdx.x & 31;
  const int row  = lane & 15;
  const int ko   = (lane >> 4) << 3;   // 0 or 8
  const bf16_t* p = base + row * ld + ko;
  v8bf lo = *(const v8bf*)(p);
  v8bf hi = *(const v8bf*)(p + 16);
  v16bf f;
#pragma unroll
  for (int i = 0; i < 8; ++i) { f[i] = lo[i]; f[i + 8] = hi[i]; }
  return f;
}

// B-operand fragment. B stored transposed (16 rows x K cols, row = output col).
// Lanes 0-15 hold K 0..7 / 16..23 of col=lane, lanes 16-31 hold K 8..15 / 24..31 of same col.
__device__ __forceinline__ v16bf load_frag_b(const bf16_t* __restrict__ base, int ld) {
  const int lane = threadIdx.x & 31;
  const int colr = lane & 15;
  const int ko   = (lane >> 4) << 3;   // 0 or 8
  const bf16_t* p = base + colr * ld + ko;
  v8bf lo = *(const v8bf*)(p);
  v8bf hi = *(const v8bf*)(p + 16);
  v16bf f;
#pragma unroll
  for (int i = 0; i < 8; ++i) { f[i] = lo[i]; f[i + 8] = hi[i]; }
  return f;
}

__device__ __forceinline__ v8f wmma_bf16(v16bf a, v16bf b, v8f c) {
  return __builtin_amdgcn_wmma_f32_16x16x32_bf16(false, a, false, b, (short)0, c,
                                                 false, false);
}

// ---------- precision conversion / weight transposition ----------

__global__ void __launch_bounds__(256) convert_bf16(const float* __restrict__ src,
                                                    bf16_t* __restrict__ dst, int n) {
  int i = blockIdx.x * blockDim.x + threadIdx.x;
  if (i < n) dst[i] = f2bf(src[i]);
}

// src is R x C row-major; dst is C x R row-major (i.e. transposed)
__global__ void __launch_bounds__(256) transpose_bf16(const float* __restrict__ src,
                                                      bf16_t* __restrict__ dst,
                                                      int R, int C) {
  int i = blockIdx.x * blockDim.x + threadIdx.x;
  if (i < R * C) {
    int c = i / R;
    int r = i - c * R;
    dst[i] = f2bf(src[(size_t)r * C + c]);
  }
}

// ---------- QKV projection: [B*S,D] x [D,3D] + bias -> q/k/vT (bf16) ----------
// grid = (B*S/128, 3D/64), block = 128 (4 waves).
// Wave w: 32x64 tile -> 6 fragment loads (one clause) + 8 WMMA per k-step.

__global__ void __launch_bounds__(128) qkv_gemm(const bf16_t* __restrict__ xb,
                                                const bf16_t* __restrict__ wT,
                                                const float* __restrict__ bias,
                                                bf16_t* __restrict__ qb,
                                                bf16_t* __restrict__ kb,
                                                bf16_t* __restrict__ vt) {
  const int wave = threadIdx.x >> 5;
  const int lane = threadIdx.x & 31;
  const int m0 = blockIdx.x * 128 + wave * 32;
  const int n0 = blockIdx.y * 64;

  v8f acc[2][4];
#pragma unroll
  for (int t = 0; t < 2; ++t)
#pragma unroll
    for (int n = 0; n < 4; ++n)
#pragma unroll
      for (int r = 0; r < 8; ++r) acc[t][n][r] = 0.f;

  const bf16_t* arow0 = xb + (size_t)m0 * DD;
  const bf16_t* arow1 = xb + (size_t)(m0 + 16) * DD;
  const bf16_t* brow0 = wT + (size_t)(n0 + 0)  * DD;
  const bf16_t* brow1 = wT + (size_t)(n0 + 16) * DD;
  const bf16_t* brow2 = wT + (size_t)(n0 + 32) * DD;
  const bf16_t* brow3 = wT + (size_t)(n0 + 48) * DD;

  for (int k0 = 0; k0 < DD; k0 += 32) {
    // distinct fragment registers -> compiler can pipeline s_wait_loadcnt
    v16bf a0 = load_frag_a(arow0 + k0, DD);
    v16bf a1 = load_frag_a(arow1 + k0, DD);
    v16bf b0 = load_frag_b(brow0 + k0, DD);
    v16bf b1 = load_frag_b(brow1 + k0, DD);
    v16bf b2 = load_frag_b(brow2 + k0, DD);
    v16bf b3 = load_frag_b(brow3 + k0, DD);
    if (k0 + 32 < DD) {  // lane-spread prefetch of next k-tile (global_prefetch_b8)
      __builtin_prefetch(arow0 + k0 + 32 + lane * 32, 0, 3);
      __builtin_prefetch(brow0 + k0 + 32 + lane * 32, 0, 3);
    }
    acc[0][0] = wmma_bf16(a0, b0, acc[0][0]);
    acc[1][0] = wmma_bf16(a1, b0, acc[1][0]);
    acc[0][1] = wmma_bf16(a0, b1, acc[0][1]);
    acc[1][1] = wmma_bf16(a1, b1, acc[1][1]);
    acc[0][2] = wmma_bf16(a0, b2, acc[0][2]);
    acc[1][2] = wmma_bf16(a1, b2, acc[1][2]);
    acc[0][3] = wmma_bf16(a0, b3, acc[0][3]);
    acc[1][3] = wmma_bf16(a1, b3, acc[1][3]);
  }

  const int col = lane & 15;
  const int rbase = (lane >> 4) << 3;
#pragma unroll
  for (int t = 0; t < 2; ++t) {
#pragma unroll
    for (int j = 0; j < 4; ++j) {
      int n = n0 + j * 16 + col;
      float bv = bias[n];
      int which = n / DD;
      int dcol = n - which * DD;
      int h = dcol / HDIM;
      int hd = dcol - h * HDIM;
#pragma unroll
      for (int r = 0; r < 8; ++r) {
        int m = m0 + t * 16 + rbase + r;
        int b_ = m / SS;
        int srow = m - b_ * SS;
        float val = acc[t][j][r] + bv;
        size_t bh = (size_t)(b_ * HH + h);
        if (which == 0)      qb[(bh * SS + srow) * HDIM + hd] = f2bf(val * SCALE_Q);
        else if (which == 1) kb[(bh * SS + srow) * HDIM + hd] = f2bf(val);
        else                 vt[(bh * HDIM + hd) * SS + srow] = f2bf(val);
      }
    }
  }
}

// ---------- flash-style causal attention ----------
// grid = (B*H, S/64), block = 128 (4 waves, each owns 16 query rows).

__global__ void __launch_bounds__(128) attn_kernel(const bf16_t* __restrict__ qbuf,
                                                   const bf16_t* __restrict__ kbuf,
                                                   const bf16_t* __restrict__ vtbuf,
                                                   bf16_t* __restrict__ yb) {
  __shared__ __align__(16) bf16_t psh[4][16 * 64];

  const int bh   = blockIdx.x;
  const int wave = threadIdx.x >> 5;
  const int lane = threadIdx.x & 31;
  const int qw   = blockIdx.y * 64 + wave * 16;   // wave's first query row

  const bf16_t* q = qbuf  + (size_t)bh * SS * HDIM;
  const bf16_t* k = kbuf  + (size_t)bh * SS * HDIM;
  const bf16_t* v = vtbuf + (size_t)bh * HDIM * SS;   // [HD][S]
  bf16_t* pw = &psh[wave][0];

  const int col = lane & 15;
  const int rbase = (lane >> 4) << 3;

  // Q fragments for K-dim 64 (2 x k32); softmax scale folded into q at store time
  v16bf qf0 = load_frag_a(q + (size_t)qw * HDIM, HDIM);
  v16bf qf1 = load_frag_a(q + (size_t)qw * HDIM + 32, HDIM);

  v8f acc[4];
  float mrow[8], lrow[8];
#pragma unroll
  for (int n = 0; n < 4; ++n)
#pragma unroll
    for (int r = 0; r < 8; ++r) acc[n][r] = 0.f;
#pragma unroll
  for (int r = 0; r < 8; ++r) { mrow[r] = -3.0e38f; lrow[r] = 0.f; }

  const int nkeys = qw + 16;                 // causal: keys 0..qw+15
  for (int k0 = 0; k0 < nkeys; k0 += 64) {
    if (k0 + 64 < nkeys)  // hint next K tile into cache
      __builtin_prefetch(k + (size_t)(k0 + 64) * HDIM + lane * 64, 0, 3);

    // ---- scores: 16 x 64, S = Q K^T  (preload all 8 K fragments) ----
    const bf16_t* kt = k + (size_t)k0 * HDIM;
    v16bf kf0 = load_frag_b(kt, HDIM);
    v16bf kf1 = load_frag_b(kt + 32, HDIM);
    v16bf kf2 = load_frag_b(kt + 16 * HDIM, HDIM);
    v16bf kf3 = load_frag_b(kt + 16 * HDIM + 32, HDIM);
    v16bf kf4 = load_frag_b(kt + 32 * HDIM, HDIM);
    v16bf kf5 = load_frag_b(kt + 32 * HDIM + 32, HDIM);
    v16bf kf6 = load_frag_b(kt + 48 * HDIM, HDIM);
    v16bf kf7 = load_frag_b(kt + 48 * HDIM + 32, HDIM);

    v8f s[4];
#pragma unroll
    for (int n = 0; n < 4; ++n)
#pragma unroll
      for (int r = 0; r < 8; ++r) s[n][r] = 0.f;
    s[0] = wmma_bf16(qf0, kf0, s[0]);
    s[0] = wmma_bf16(qf1, kf1, s[0]);
    s[1] = wmma_bf16(qf0, kf2, s[1]);
    s[1] = wmma_bf16(qf1, kf3, s[1]);
    s[2] = wmma_bf16(qf0, kf4, s[2]);
    s[2] = wmma_bf16(qf1, kf5, s[2]);
    s[3] = wmma_bf16(qf0, kf6, s[3]);
    s[3] = wmma_bf16(qf1, kf7, s[3]);

    // ---- causal mask (only tiles crossing the diagonal) ----
    if (k0 + 63 > qw) {
#pragma unroll
      for (int j = 0; j < 4; ++j) {
        int key = k0 + j * 16 + col;
#pragma unroll
        for (int r = 0; r < 8; ++r)
          if (key > qw + rbase + r) s[j][r] = -3.0e38f;
      }
    }
    // ---- online softmax: row max (16-lane xor reduce == row group) ----
    float tmax[8];
#pragma unroll
    for (int r = 0; r < 8; ++r)
      tmax[r] = fmaxf(fmaxf(s[0][r], s[1][r]), fmaxf(s[2][r], s[3][r]));
#pragma unroll
    for (int msk = 1; msk <= 8; msk <<= 1)
#pragma unroll
      for (int r = 0; r < 8; ++r)
        tmax[r] = fmaxf(tmax[r], __shfl_xor(tmax[r], msk, 32));
#pragma unroll
    for (int r = 0; r < 8; ++r) {
      float mnew = fmaxf(mrow[r], tmax[r]);
      float sc = __expf(mrow[r] - mnew);
      mrow[r] = mnew;
      lrow[r] *= sc;
#pragma unroll
      for (int n = 0; n < 4; ++n) acc[n][r] *= sc;
    }
    // ---- P = exp(s-m), stage to LDS (C-layout -> A-layout re-tile) ----
    float psum[8];
#pragma unroll
    for (int r = 0; r < 8; ++r) psum[r] = 0.f;
#pragma unroll
    for (int j = 0; j < 4; ++j)
#pragma unroll
      for (int r = 0; r < 8; ++r) {
        float p = __expf(s[j][r] - mrow[r]);
        psum[r] += p;
        pw[(rbase + r) * 64 + j * 16 + col] = f2bf(p);
      }
#pragma unroll
    for (int msk = 1; msk <= 8; msk <<= 1)
#pragma unroll
      for (int r = 0; r < 8; ++r) psum[r] += __shfl_xor(psum[r], msk, 32);
#pragma unroll
    for (int r = 0; r < 8; ++r) lrow[r] += psum[r];

    // ---- acc += P @ V (preload P from LDS + all 8 V fragments) ----
    v16bf pf0 = load_frag_a(pw, 64);
    v16bf pf1 = load_frag_a(pw + 32, 64);
    const bf16_t* vt0 = v + k0;
    v16bf vf0 = load_frag_b(vt0, SS);
    v16bf vf1 = load_frag_b(vt0 + 32, SS);
    v16bf vf2 = load_frag_b(vt0 + (size_t)16 * SS, SS);
    v16bf vf3 = load_frag_b(vt0 + (size_t)16 * SS + 32, SS);
    v16bf vf4 = load_frag_b(vt0 + (size_t)32 * SS, SS);
    v16bf vf5 = load_frag_b(vt0 + (size_t)32 * SS + 32, SS);
    v16bf vf6 = load_frag_b(vt0 + (size_t)48 * SS, SS);
    v16bf vf7 = load_frag_b(vt0 + (size_t)48 * SS + 32, SS);
    acc[0] = wmma_bf16(pf0, vf0, acc[0]);
    acc[0] = wmma_bf16(pf1, vf1, acc[0]);
    acc[1] = wmma_bf16(pf0, vf2, acc[1]);
    acc[1] = wmma_bf16(pf1, vf3, acc[1]);
    acc[2] = wmma_bf16(pf0, vf4, acc[2]);
    acc[2] = wmma_bf16(pf1, vf5, acc[2]);
    acc[3] = wmma_bf16(pf0, vf6, acc[3]);
    acc[3] = wmma_bf16(pf1, vf7, acc[3]);
  }

  // ---- normalize, store y as bf16 in [B,S,D] (heads re-interleaved) ----
  const int b_ = bh / HH;
  const int h  = bh - b_ * HH;
#pragma unroll
  for (int r = 0; r < 8; ++r) {
    float inv = 1.0f / lrow[r];
    int srow = qw + rbase + r;
    size_t rowoff = ((size_t)(b_ * SS + srow)) * DD + h * HDIM;
#pragma unroll
    for (int n = 0; n < 4; ++n)
      yb[rowoff + n * 16 + col] = f2bf(acc[n][r] * inv);
  }
}

// ---------- output projection: [B*S,D] x [D,D] + bias -> fp32 out ----------
// grid = (B*S/128, D/64), block = 128 (4 waves), 32x64 tile per wave.

__global__ void __launch_bounds__(128) proj_gemm(const bf16_t* __restrict__ yb,
                                                 const bf16_t* __restrict__ wT,
                                                 const float* __restrict__ bias,
                                                 float* __restrict__ out) {
  const int wave = threadIdx.x >> 5;
  const int lane = threadIdx.x & 31;
  const int m0 = blockIdx.x * 128 + wave * 32;
  const int n0 = blockIdx.y * 64;

  v8f acc[2][4];
#pragma unroll
  for (int t = 0; t < 2; ++t)
#pragma unroll
    for (int n = 0; n < 4; ++n)
#pragma unroll
      for (int r = 0; r < 8; ++r) acc[t][n][r] = 0.f;

  const bf16_t* arow0 = yb + (size_t)m0 * DD;
  const bf16_t* arow1 = yb + (size_t)(m0 + 16) * DD;
  const bf16_t* brow0 = wT + (size_t)(n0 + 0)  * DD;
  const bf16_t* brow1 = wT + (size_t)(n0 + 16) * DD;
  const bf16_t* brow2 = wT + (size_t)(n0 + 32) * DD;
  const bf16_t* brow3 = wT + (size_t)(n0 + 48) * DD;

  for (int k0 = 0; k0 < DD; k0 += 32) {
    v16bf a0 = load_frag_a(arow0 + k0, DD);
    v16bf a1 = load_frag_a(arow1 + k0, DD);
    v16bf b0 = load_frag_b(brow0 + k0, DD);
    v16bf b1 = load_frag_b(brow1 + k0, DD);
    v16bf b2 = load_frag_b(brow2 + k0, DD);
    v16bf b3 = load_frag_b(brow3 + k0, DD);
    if (k0 + 32 < DD) {
      __builtin_prefetch(arow0 + k0 + 32 + lane * 32, 0, 3);
      __builtin_prefetch(brow0 + k0 + 32 + lane * 32, 0, 3);
    }
    acc[0][0] = wmma_bf16(a0, b0, acc[0][0]);
    acc[1][0] = wmma_bf16(a1, b0, acc[1][0]);
    acc[0][1] = wmma_bf16(a0, b1, acc[0][1]);
    acc[1][1] = wmma_bf16(a1, b1, acc[1][1]);
    acc[0][2] = wmma_bf16(a0, b2, acc[0][2]);
    acc[1][2] = wmma_bf16(a1, b2, acc[1][2]);
    acc[0][3] = wmma_bf16(a0, b3, acc[0][3]);
    acc[1][3] = wmma_bf16(a1, b3, acc[1][3]);
  }

  const int col = lane & 15;
  const int rbase = (lane >> 4) << 3;
#pragma unroll
  for (int t = 0; t < 2; ++t)
#pragma unroll
    for (int j = 0; j < 4; ++j) {
      int n = n0 + j * 16 + col;
      float bv = bias[n];
#pragma unroll
      for (int r = 0; r < 8; ++r) {
        int m = m0 + t * 16 + rbase + r;
        out[(size_t)m * DD + n] = acc[t][j][r] + bv;
      }
    }
}

// ---------- launcher ----------

extern "C" void kernel_launch(void* const* d_in, const int* in_sizes, int n_in,
                              void* d_out, int out_size, void* d_ws, size_t ws_size,
                              hipStream_t stream) {
  const float* x      = (const float*)d_in[0];
  const float* w_qkv  = (const float*)d_in[1];
  const float* b_qkv  = (const float*)d_in[2];
  const float* w_proj = (const float*)d_in[3];
  const float* b_proj = (const float*)d_in[4];
  float* out = (float*)d_out;

  // workspace layout (bytes): xb 16M | wqkvT 6M | wprojT 2M | q 16M | k 16M | vT 16M | y 16M
  char* ws = (char*)d_ws;
  bf16_t* xb     = (bf16_t*)(ws);
  bf16_t* wqkvT  = (bf16_t*)(ws + (16u << 20));
  bf16_t* wprojT = (bf16_t*)(ws + (22u << 20));
  bf16_t* qb     = (bf16_t*)(ws + (24u << 20));
  bf16_t* kb     = (bf16_t*)(ws + (40u << 20));
  bf16_t* vt     = (bf16_t*)(ws + (56u << 20));
  bf16_t* yb     = (bf16_t*)(ws + (72u << 20));

  const int nx = BB * SS * DD;
  convert_bf16<<<(nx + 255) / 256, 256, 0, stream>>>(x, xb, nx);
  const int nw1 = DD * 3 * DD;
  transpose_bf16<<<(nw1 + 255) / 256, 256, 0, stream>>>(w_qkv, wqkvT, DD, 3 * DD);
  const int nw2 = DD * DD;
  transpose_bf16<<<(nw2 + 255) / 256, 256, 0, stream>>>(w_proj, wprojT, DD, DD);

  dim3 g1(BB * SS / 128, 3 * DD / 64);
  qkv_gemm<<<g1, 128, 0, stream>>>(xb, wqkvT, b_qkv, qb, kb, vt);

  dim3 g2(BB * HH, SS / 64);
  attn_kernel<<<g2, 128, 0, stream>>>(qb, kb, vt, yb);

  dim3 g3(BB * SS / 128, DD / 64);
  proj_gemm<<<g3, 128, 0, stream>>>(yb, wprojT, b_proj, out);
}
